// Bonsai_5257039970324
// MI455X (gfx1250) — matrix operand order repaired and checked by reading, not compiled, not run on hardware
//
#include <hip/hip_runtime.h>
#include <hip/hip_bf16.h>
#include <stdint.h>

#define NF 512
#define PD 64
#define NC 32
#define INT_N 15
#define TOT_N 31
#define SIGC 4.0f

typedef _Float16 half_t;
typedef __attribute__((ext_vector_type(16))) _Float16 v16h;
typedef __attribute__((ext_vector_type(8)))  float    v8f;
typedef __attribute__((ext_vector_type(4)))  unsigned int u32x4;
typedef __attribute__((ext_vector_type(8)))  int      i32x8;
typedef __attribute__((ext_vector_type(4)))  int      i32x4;

// f16 fragment blob layout in d_ws (halves):
//   [0, 126976)          : V,W B-fragments  [m:2][n:31][t:2][s:2][lane:32][e:16]
//   [126976, 128000)     : T^T B-fragments  [s:2][lane:32][e:16]
//   [128000, 160768)     : Z   B-fragments  [s:16][t:4][lane:32][e:16]
#define WV_HALVES 126976
#define T_HALVES  1024
#define Z_HALVES  32768
#define ZOFF      128000
#define WVT_BYTES 256000

// dynamic LDS map (bytes)
#define PP_OFF 256000      // [8 waves][16 rows][64] f32
#define TB_OFF 288768      // [8 waves][16 rows][16 nodes] f32
#define I_OFF  296960      // [8 waves][31 nodes][16 rows] f32
#define LDS_BYTES 312832

#define FRAG_OFF(m,n,t,s) (((((m)*TOT_N + (n))*2 + (t))*2 + (s))*512)

__device__ __forceinline__ v8f wmma16(v16h a, v16h b, v8f c) {
  return __builtin_amdgcn_wmma_f32_16x16x32_f16(false, a, false, b, (short)0, c,
                                                false, false);
}

__device__ __forceinline__ float fast_tanh(float v) {
  // tanh(v) = 1 - 2/(e^{2v}+1); v_exp_f32 + v_rcp_f32
  float e = __expf(2.0f * v);
  return 1.0f - 2.0f * __builtin_amdgcn_rcpf(e + 1.0f);
}

// ---------------- prep: f32 -> f16 fragment layouts in d_ws ----------------
__global__ void bonsai_prep(const float* __restrict__ Z, const float* __restrict__ T,
                            const float* __restrict__ V, const float* __restrict__ W,
                            half_t* __restrict__ ws) {
  int idx = blockIdx.x * blockDim.x + threadIdx.x;
  if (idx < WV_HALVES) {
    int e    = idx & 15;
    int lane = (idx >> 4) & 31;
    int s    = (idx >> 9) & 1;
    int t    = (idx >> 10) & 1;
    int n    = (idx >> 11) % TOT_N;
    int m    = (idx >> 11) / TOT_N;      // 0 = V, 1 = W
    int K    = 32 * s + 16 * (lane >> 4) + e;
    int col  = 16 * t + (lane & 15);
    const float* M = (m == 0) ? V : W;
    ws[idx] = (half_t)M[((size_t)n * PD + K) * NC + col];
  } else if (idx < WV_HALVES + T_HALVES) {
    int rel  = idx - WV_HALVES;
    int e    = rel & 15;
    int lane = (rel >> 4) & 31;
    int s    = rel >> 9;
    int K    = 32 * s + 16 * (lane >> 4) + e;
    int col  = lane & 15;
    ws[idx] = (col < INT_N) ? (half_t)T[col * PD + K] : (half_t)0.0f;
  } else if (idx < WV_HALVES + T_HALVES + Z_HALVES) {
    int rel  = idx - ZOFF;
    int e    = rel & 15;
    int lane = (rel >> 4) & 31;
    int t    = (rel >> 9) & 3;
    int s    = rel >> 11;
    int K    = 32 * s + 16 * (lane >> 4) + e;
    int col  = 16 * t + (lane & 15);
    ws[idx] = (half_t)Z[K * PD + col];
  }
}

// ---------------- fused main kernel ----------------
__global__ void bonsai_main(const float* __restrict__ x,
                            const half_t* __restrict__ ws,
                            float* __restrict__ out, int batch) {
  extern __shared__ __align__(128) char smem[];
  half_t* ldsWVT = (half_t*)smem;
  float*  shPP   = (float*)(smem + PP_OFF);
  float*  shTB   = (float*)(smem + TB_OFF);
  float*  shI    = (float*)(smem + I_OFF);

  const int lane = threadIdx.x & 31;
  const int wave = threadIdx.x >> 5;
  const int mrow = lane & 15;
  const int lhi  = lane >> 4;
  const v8f vzero = {0.f,0.f,0.f,0.f,0.f,0.f,0.f,0.f};

  // --- stage 256000-byte WV+T fragment blob into LDS via Tensor Data Mover ---
  if (wave == 0) {
    unsigned long long ga = (unsigned long long)(uintptr_t)ws;
    u32x4 g0;
    g0[0] = 1u;                                    // count=1, gather off
    g0[1] = __builtin_amdgcn_groupstaticsize();    // lds_addr (dynamic LDS base)
    g0[2] = (unsigned)(ga & 0xffffffffull);
    g0[3] = (unsigned)((ga >> 32) & 0x01ffffffull) | (2u << 30); // addr hi | type=2
    const unsigned NE = WVT_BYTES / 8;             // 32000 8-byte elements
    i32x8 g1;
    g1[0] = (int)(3u << 16);                       // data_size = 8B
    g1[1] = (int)((NE & 0xffffu) << 16);           // tensor_dim0 lo16
    g1[2] = (int)(((NE >> 16) & 0xffffu) | (1u << 16)); // dim0 hi | tensor_dim1=1
    g1[3] = (int)((NE & 0xffffu) << 16);           // tile_dim0
    g1[4] = 1;                                     // tile_dim1 = 1
    g1[5] = (int)NE;                               // tensor_dim0_stride
    g1[6] = 0;
    g1[7] = 0;
    i32x4 g2 = {0,0,0,0}, g3 = {0,0,0,0};
    i32x8 g4 = {0,0,0,0,0,0,0,0};
    __builtin_amdgcn_tensor_load_to_lds(g0, g1, g2, g3, g4, 0);
    __builtin_amdgcn_s_wait_tensorcnt(0);
  }
  __syncthreads();

  const int numTiles = (batch + 15) >> 4;
  const int chunks   = (numTiles + 7) >> 3;

  for (int chunk = blockIdx.x; chunk < chunks; chunk += gridDim.x) {
    const int tile = chunk * 8 + wave;
    if (tile >= numTiles) continue;            // wave-uniform; no barriers in loop
    const int row0 = tile * 16;

    // ---------- GEMM1: pp = (x @ Z) / PD, 16x64 tile ----------
    v8f ppAcc[4];
    #pragma unroll
    for (int t = 0; t < 4; ++t) ppAcc[t] = vzero;

    // Branchless: clamp row so out-of-range rows read row batch-1 (their
    // results are never stored). Keeps EXEC full and lets loads pipeline.
    int myrow = row0 + mrow;
    if (myrow >= batch) myrow = batch - 1;
    const float* xr = x + (size_t)myrow * NF;

    for (int s = 0; s < 16; ++s) {
      const int kb = 32 * s + 16 * lhi;
      // load all 4 B fragments first -> one big clause, staggered waits
      v16h bf[4];
      #pragma unroll
      for (int t = 0; t < 4; ++t)
        bf[t] = *(const v16h*)(ws + ZOFF + ((size_t)(s * 4 + t) * 32 + lane) * 16);
      v16h a;
      #pragma unroll
      for (int e = 0; e < 8; ++e) {
        a[e]     = (half_t)xr[kb + e];
        a[8 + e] = (half_t)xr[kb + 8 + e];
      }
      #pragma unroll
      for (int t = 0; t < 4; ++t) ppAcc[t] = wmma16(a, bf[t], ppAcc[t]);
    }

    // stage scaled pp tile to LDS (C-layout -> row major [16][64])
    float* ppw = shPP + wave * (16 * 64);
    #pragma unroll
    for (int t = 0; t < 4; ++t)
      #pragma unroll
      for (int r = 0; r < 8; ++r)
        ppw[(r + 8 * lhi) * 64 + (16 * t + mrow)] = ppAcc[t][r] * (1.0f / PD);

    // ---------- rebuild pp as A-fragments (f16) ----------
    v16h pA[2];
    {
      const float* pr = ppw + mrow * 64;
      #pragma unroll
      for (int s2 = 0; s2 < 2; ++s2) {
        const int kb = 32 * s2 + 16 * lhi;
        v16h a;
        #pragma unroll
        for (int e = 0; e < 8; ++e) {
          a[e]     = (half_t)pr[kb + e];
          a[8 + e] = (half_t)pr[kb + 8 + e];
        }
        pA[s2] = a;
      }
    }

    // ---------- tb = tanh(SIG * pp @ T^T) ----------
    {
      v16h tb0 = *(const v16h*)(ldsWVT + WV_HALVES + 0 * 512 + lane * 16);
      v16h tb1 = *(const v16h*)(ldsWVT + WV_HALVES + 1 * 512 + lane * 16);
      v8f tAcc = wmma16(pA[1], tb1, wmma16(pA[0], tb0, vzero));
      float* tbw = shTB + wave * 256;
      #pragma unroll
      for (int r = 0; r < 8; ++r)
        tbw[(r + 8 * lhi) * 16 + mrow] = fast_tanh(SIGC * tAcc[r]);
    }

    // ---------- tree indicators (lanes 0..15, one row each) ----------
    if (lane < 16) {
      const float* tbw = shTB + wave * 256 + lane * 16;
      float tbl[INT_N];
      #pragma unroll
      for (int j = 0; j < INT_N; ++j) tbl[j] = tbw[j];
      float I[TOT_N];
      I[0] = 1.0f;
      #pragma unroll
      for (int i = 1; i < TOT_N; ++i) {
        int j = (i + 1) / 2 - 1;
        float sgn = (i & 1) ? 1.0f : -1.0f;   // left child: +, right child: -
        I[i] = 0.5f * I[j] * (1.0f + sgn * tbl[j]);
      }
      float* Iw = shI + wave * (TOT_N * 16);
      #pragma unroll
      for (int n = 0; n < TOT_N; ++n) Iw[n * 16 + lane] = I[n];
    }

    // ---------- per-node GEMMs + epilogue ----------
    v8f sAcc[2];
    sAcc[0] = vzero; sAcc[1] = vzero;
    const float* Iw = shI + wave * (TOT_N * 16);

    #pragma unroll 2
    for (int n = 0; n < TOT_N; ++n) {
      float Irows[8];
      const float* ip = Iw + n * 16 + 8 * lhi;   // rows r+8*lhi, consecutive
      #pragma unroll
      for (int r = 0; r < 8; ++r) Irows[r] = ip[r];

      #pragma unroll
      for (int t2 = 0; t2 < 2; ++t2) {
        // bulk-load all four fragments, then do the WMMA group
        v16h bv0 = *(const v16h*)(ldsWVT + FRAG_OFF(0, n, t2, 0) + lane * 16);
        v16h bv1 = *(const v16h*)(ldsWVT + FRAG_OFF(0, n, t2, 1) + lane * 16);
        v16h bw0 = *(const v16h*)(ldsWVT + FRAG_OFF(1, n, t2, 0) + lane * 16);
        v16h bw1 = *(const v16h*)(ldsWVT + FRAG_OFF(1, n, t2, 1) + lane * 16);
        v8f va = wmma16(pA[1], bv1, wmma16(pA[0], bv0, vzero));
        v8f wa = wmma16(pA[1], bw1, wmma16(pA[0], bw0, vzero));
        #pragma unroll
        for (int r = 0; r < 8; ++r)
          sAcc[t2][r] += wa[r] * fast_tanh(SIGC * va[r]) * Irows[r];
      }
    }

    // ---------- store score [16 x 32] ----------
    #pragma unroll
    for (int t2 = 0; t2 < 2; ++t2)
      #pragma unroll
      for (int r = 0; r < 8; ++r) {
        int row = row0 + r + 8 * lhi;
        if (row < batch) out[(size_t)row * NC + 16 * t2 + mrow] = sAcc[t2][r];
      }
  }
}

extern "C" void kernel_launch(void* const* d_in, const int* in_sizes, int n_in,
                              void* d_out, int out_size, void* d_ws, size_t ws_size,
                              hipStream_t stream) {
  (void)n_in; (void)out_size; (void)ws_size;
  const float* x = (const float*)d_in[0];
  const float* Z = (const float*)d_in[1];
  const float* T = (const float*)d_in[2];
  const float* V = (const float*)d_in[3];
  const float* W = (const float*)d_in[4];
  float* out = (float*)d_out;
  half_t* ws = (half_t*)d_ws;

  const int batch = in_sizes[0] / NF;

  const int totalHalves = WV_HALVES + T_HALVES + Z_HALVES;
  bonsai_prep<<<(totalHalves + 255) / 256, 256, 0, stream>>>(Z, T, V, W, ws);

  const int numTiles = (batch + 15) / 16;
  const int chunks   = (numTiles + 7) / 8;
  const int nwg      = chunks < 160 ? chunks : 160;   // ~1 persistent WG per WGP
  bonsai_main<<<nwg, 256, LDS_BYTES, stream>>>(x, ws, out, batch);
}
